// Attention_29618094473466
// MI455X (gfx1250) — matrix operand order repaired and checked by reading, not compiled, not run on hardware
//
#include <hip/hip_runtime.h>
#include <math.h>

// Problem constants (match reference)
#define BATCH 16
#define TT_   512
#define TS_   512
#define H_    1024
#define O_    1024

// GEMM tiling: 256 threads = 8 wave32, wave grid 4(M) x 2(N), wave tile 32x32
#define BM 128
#define BN 64
#define BK 64

typedef __attribute__((ext_vector_type(16))) __bf16 v16bf;
typedef __attribute__((ext_vector_type(8)))  __bf16 v8bf;
typedef __attribute__((ext_vector_type(8)))  float  v8f;

// ---------------------------------------------------------------------------
// CDNA5 async global->LDS (GLOBAL_LOAD_ASYNC_TO_LDS_B128, ASYNCcnt-tracked)
// Builtin signature on this toolchain (from hipcc diagnostics): both pointer
// params are plain generic `int __attribute__((vector_size(16)))*`, then
// imm offset + imm cpol.
// ---------------------------------------------------------------------------
#if defined(__gfx1250__) && __has_builtin(__builtin_amdgcn_global_load_async_to_lds_b128)
#define HAVE_ASYNC 1
#else
#define HAVE_ASYNC 0
#endif

#if HAVE_ASYNC
typedef int v4i_gcc __attribute__((vector_size(16)));
__device__ __forceinline__ void async_b128(const __bf16* gsrc, __bf16* ldst) {
  __builtin_amdgcn_global_load_async_to_lds_b128(
      (v4i_gcc*)gsrc, (v4i_gcc*)ldst, 0, 0);
}
#endif

__device__ __forceinline__ void wait_async0() {
#if defined(__gfx1250__)
#if __has_builtin(__builtin_amdgcn_s_wait_asynccnt)
  __builtin_amdgcn_s_wait_asynccnt(0);
#elif HAVE_ASYNC
  asm volatile("s_wait_asynccnt 0x0" ::: "memory");
#endif
#endif
}

// ---------------------------------------------------------------------------
// fp32 -> bf16 cast
// ---------------------------------------------------------------------------
__global__ void cvt_f32_bf16(const float* __restrict__ x, __bf16* __restrict__ y, int n) {
  int i = blockIdx.x * blockDim.x + threadIdx.x;
  if (i < n) y[i] = (__bf16)x[i];
}

// ---------------------------------------------------------------------------
// Tiled fp32 -> bf16 cast + transpose: in [R][C] f32  ->  out [C][R] bf16
// ---------------------------------------------------------------------------
__global__ __launch_bounds__(256) void transpose_cast(const float* __restrict__ in,
                                                      __bf16* __restrict__ out,
                                                      int R, int C,
                                                      long long sIn, long long sOut) {
  __shared__ __bf16 tile[32][33];
  const int bz = blockIdx.z;
  const float* ib = in + (size_t)bz * sIn;
  __bf16* ob = out + (size_t)bz * sOut;
  const int c0 = blockIdx.x * 32;
  const int r0 = blockIdx.y * 32;
  const int tx = threadIdx.x & 31;
  const int ty = threadIdx.x >> 5;  // 0..7
#pragma unroll
  for (int i = 0; i < 4; ++i) {
    int r = ty + 8 * i;
    tile[r][tx] = (__bf16)ib[(size_t)(r0 + r) * C + (c0 + tx)];
  }
  __syncthreads();
#pragma unroll
  for (int i = 0; i < 4; ++i) {
    int r = ty + 8 * i;  // output row c0+r, output col r0+tx (coalesced)
    ob[(size_t)(c0 + r) * R + (r0 + tx)] = tile[tx][r];
  }
}

// ---------------------------------------------------------------------------
// Fragment loaders (WMMA 16x16x32 bf16 VGPR layouts, ISA 7.12.2); ld = BK
// ---------------------------------------------------------------------------
__device__ __forceinline__ v16bf load_a_frag(const __bf16* S, int row0, int k0, int lane) {
  int row = row0 + (lane & 15);
  int kb  = k0 + (lane >> 4) * 8;
  v8bf lo = *(const v8bf*)(S + row * BK + kb);        // K kb..kb+7
  v8bf hi = *(const v8bf*)(S + row * BK + kb + 16);   // K kb+16..kb+23
  v16bf r;
#pragma unroll
  for (int i = 0; i < 8; ++i) { r[i] = lo[i]; r[i + 8] = hi[i]; }
  return r;
}

__device__ __forceinline__ v16bf load_b_frag(const __bf16* S, int n0, int k0, int lane) {
  int n  = n0 + (lane & 15);
  int kb = k0 + (lane >> 4) * 8;
  v8bf lo = *(const v8bf*)(S + n * BK + kb);
  v8bf hi = *(const v8bf*)(S + n * BK + kb + 16);
  v16bf r;
#pragma unroll
  for (int i = 0; i < 8; ++i) { r[i] = lo[i]; r[i + 8] = hi[i]; }
  return r;
}

// ---------------------------------------------------------------------------
// Batched bf16 WMMA GEMM: C[bz] = epi(A[bz] @ B_stored[bz]^T)
// Both A and B_stored are [rows][K] with K contiguous (ldX = K-stride).
//   EPI    : 0 = store bf16, 1 = store f32, 2 = tanh(x + bias) f32
//   CONCAT : A is concat(A0,A1) along K at k = ksplit
// Double-buffered LDS; async global->LDS staging when available.
// ---------------------------------------------------------------------------
struct GArgs {
  const __bf16* A0; const __bf16* A1; long long sA; int lda;
  const __bf16* B;  long long sB; int ldb;
  void* C; long long sC; int ldc;
  const float* bias;
  int K, ksplit;
};

template <int EPI, bool CONCAT>
__global__ __launch_bounds__(256) void wgemm_bf16(GArgs g) {
  __shared__ __bf16 Asb[2][BM * BK];  // 2 x 16 KB
  __shared__ __bf16 Bsb[2][BN * BK];  // 2 x  8 KB

  const int bz = blockIdx.z;
  const int m0 = blockIdx.y * BM;
  const int n0 = blockIdx.x * BN;

  const __bf16* Ab0 = g.A0 + (size_t)bz * g.sA;
  const __bf16* Ab1 = CONCAT ? (g.A1 + (size_t)bz * g.sA) : nullptr;
  const __bf16* Bb  = g.B  + (size_t)bz * g.sB;

  const int tid  = threadIdx.x;
  const int lane = tid & 31;
  const int wid  = tid >> 5;
  const int mW = (wid & 3) * 32;   // wave M offset in block tile
  const int nW = (wid >> 2) * 32;  // wave N offset in block tile

  // Per-thread staging coords (16B vectors): A = 4 vecs, B = 2 vecs
  int arow[4], acol[4];
#pragma unroll
  for (int v = 0; v < 4; ++v) {
    int id = tid + 256 * v;
    arow[v] = id >> 3;        // 0..127
    acol[v] = (id & 7) * 8;   // 0..56
  }
  int brow[2], bcol[2];
#pragma unroll
  for (int v = 0; v < 2; ++v) {
    int id = tid + 256 * v;
    brow[v] = id >> 3;        // 0..63
    bcol[v] = (id & 7) * 8;
  }

#if !HAVE_ASYNC
  v8bf ra[4], rb[2];
#endif

  // Issue global reads of K-tile kt into LDS buffer `buf`
  auto issue = [&](int kt, int buf) {
    const __bf16* Ap = Ab0;
    int kk = kt * BK;
    if (CONCAT && kk >= g.ksplit) { Ap = Ab1; kk -= g.ksplit; }
#pragma unroll
    for (int v = 0; v < 4; ++v) {
      const __bf16* src = Ap + (size_t)(m0 + arow[v]) * g.lda + kk + acol[v];
#if HAVE_ASYNC
      async_b128(src, &Asb[buf][arow[v] * BK + acol[v]]);
#else
      ra[v] = *(const v8bf*)src;
#endif
    }
#pragma unroll
    for (int v = 0; v < 2; ++v) {
      const __bf16* src = Bb + (size_t)(n0 + brow[v]) * g.ldb + kt * BK + bcol[v];
#if HAVE_ASYNC
      async_b128(src, &Bsb[buf][brow[v] * BK + bcol[v]]);
#else
      rb[v] = *(const v8bf*)src;
#endif
    }
  };
  // Make staged data LDS-resident (fallback: commit regs; async: drain ASYNCcnt)
  auto commit = [&](int buf) {
#if HAVE_ASYNC
    (void)buf;
    wait_async0();
#else
#pragma unroll
    for (int v = 0; v < 4; ++v) *(v8bf*)&Asb[buf][arow[v] * BK + acol[v]] = ra[v];
#pragma unroll
    for (int v = 0; v < 2; ++v) *(v8bf*)&Bsb[buf][brow[v] * BK + bcol[v]] = rb[v];
#endif
  };

  v8f acc[2][2] = {};
  const int nk = g.K / BK;

  issue(0, 0);
  commit(0);
  __syncthreads();

  for (int kt = 0; kt < nk; ++kt) {
    const int cur = kt & 1;
    if (kt + 1 < nk) issue(kt + 1, cur ^ 1);  // overlap next stage with compute

    const __bf16* As = Asb[cur];
    const __bf16* Bs = Bsb[cur];
#pragma unroll
    for (int ks = 0; ks < BK; ks += 32) {
      v16bf a0 = load_a_frag(As, mW, ks, lane);
      v16bf a1 = load_a_frag(As, mW + 16, ks, lane);
      v16bf b0 = load_b_frag(Bs, nW, ks, lane);
      v16bf b1 = load_b_frag(Bs, nW + 16, ks, lane);
      acc[0][0] = __builtin_amdgcn_wmma_f32_16x16x32_bf16(false, a0, false, b0, (short)0,
                                                          acc[0][0], false, false);
      acc[0][1] = __builtin_amdgcn_wmma_f32_16x16x32_bf16(false, a0, false, b1, (short)0,
                                                          acc[0][1], false, false);
      acc[1][0] = __builtin_amdgcn_wmma_f32_16x16x32_bf16(false, a1, false, b0, (short)0,
                                                          acc[1][0], false, false);
      acc[1][1] = __builtin_amdgcn_wmma_f32_16x16x32_bf16(false, a1, false, b1, (short)0,
                                                          acc[1][1], false, false);
    }

    if (kt + 1 < nk) {
      commit(cur ^ 1);
      __syncthreads();
    }
  }

  // Epilogue: C/D layout VGPR r -> M = r + 8*(lane>>4), N = lane&15
  const int rBase = (lane >> 4) * 8;
  const int cBase = lane & 15;
#pragma unroll
  for (int i = 0; i < 2; ++i) {
#pragma unroll
    for (int j = 0; j < 2; ++j) {
      const int grow0 = m0 + mW + 16 * i + rBase;
      const int gcol  = n0 + nW + 16 * j + cBase;
#pragma unroll
      for (int r = 0; r < 8; ++r) {
        size_t idx = (size_t)bz * g.sC + (size_t)(grow0 + r) * g.ldc + gcol;
        float x = acc[i][j][r];
        if (EPI == 0)      ((__bf16*)g.C)[idx] = (__bf16)x;
        else if (EPI == 1) ((float*)g.C)[idx] = x;
        else               ((float*)g.C)[idx] = tanhf(x + g.bias[gcol]);
      }
    }
  }
}

// ---------------------------------------------------------------------------
// Row softmax with length mask (mirrors reference), emits bf16 weights
// ---------------------------------------------------------------------------
__global__ __launch_bounds__(256) void softmax_mask_kernel(const float* __restrict__ score,
                                                           const int* __restrict__ source,
                                                           __bf16* __restrict__ a_out) {
  const int b = blockIdx.y;
  const int t = blockIdx.x;
  const float* row = score + ((size_t)b * TT_ + t) * TS_;
  __shared__ float red[256];
  __shared__ int ired[256];
  const int tid = threadIdx.x;

  int cnt = 0;
  for (int i = tid; i < TS_; i += 256) cnt += (source[b * TS_ + i] != 0) ? 1 : 0;
  ired[tid] = cnt;
  __syncthreads();
  for (int s = 128; s > 0; s >>= 1) {
    if (tid < s) ired[tid] += ired[tid + s];
    __syncthreads();
  }
  const int lens = ired[0];

  float v0 = row[tid];
  float v1 = row[tid + 256];
  red[tid] = fmaxf(v0, v1);
  __syncthreads();
  for (int s = 128; s > 0; s >>= 1) {
    if (tid < s) red[tid] = fmaxf(red[tid], red[tid + s]);
    __syncthreads();
  }
  const float m = red[0];
  __syncthreads();

  float e0 = __expf(v0 - m) * ((tid < lens) ? 1.0f : 0.0f);
  float e1 = __expf(v1 - m) * ((tid + 256 < lens) ? 1.0f : 0.0f);
  red[tid] = e0 + e1;
  __syncthreads();
  for (int s = 128; s > 0; s >>= 1) {
    if (tid < s) red[tid] += red[tid + s];
    __syncthreads();
  }
  const float inv = 1.0f / red[0];

  __bf16* orow = a_out + ((size_t)b * TT_ + t) * TS_;
  orow[tid]       = (__bf16)(e0 * inv);
  orow[tid + 256] = (__bf16)(e1 * inv);
}

// ---------------------------------------------------------------------------
// Host-side orchestration
// ---------------------------------------------------------------------------
extern "C" void kernel_launch(void* const* d_in, const int* in_sizes, int n_in,
                              void* d_out, int out_size, void* d_ws, size_t ws_size,
                              hipStream_t stream) {
  const float* ht   = (const float*)d_in[0];  // [B,TT,H]
  const float* hs   = (const float*)d_in[1];  // [B,TS,H]
  const int*   src  = (const int*)d_in[2];    // [B,TS]
  const float* Wa   = (const float*)d_in[3];  // [H,H]
  const float* Wc   = (const float*)d_in[4];  // [2H,O]
  const float* bias = (const float*)d_in[5];  // [O]
  float* out = (float*)d_out;                 // [B,TT,O] fp32

  size_t off = 0;
  auto carve = [&](size_t bytes) {
    char* p = (char*)d_ws + off;
    off += (bytes + 255) & ~(size_t)255;
    return p;
  };
  __bf16* ht_b   = (__bf16*)carve((size_t)BATCH * TT_ * H_ * 2);
  __bf16* hs_b   = (__bf16*)carve((size_t)BATCH * TS_ * H_ * 2);
  __bf16* hs_t   = (__bf16*)carve((size_t)BATCH * H_ * TS_ * 2);  // [B,H,TS]
  __bf16* Wa_t   = (__bf16*)carve((size_t)H_ * H_ * 2);           // [N=H][K=H]
  __bf16* Wc_t   = (__bf16*)carve((size_t)O_ * 2 * H_ * 2);       // [O][2H]
  __bf16* keys_b = (__bf16*)carve((size_t)BATCH * TS_ * H_ * 2);
  float*  sc_f   = (float*)carve((size_t)BATCH * TT_ * TS_ * 4);
  __bf16* a_b    = (__bf16*)carve((size_t)BATCH * TT_ * TS_ * 2);
  __bf16* c_b    = (__bf16*)carve((size_t)BATCH * TT_ * H_ * 2);
  (void)ws_size; (void)in_sizes; (void)n_in; (void)out_size;

  // 1) casts and cast+transposes
  {
    int n = BATCH * TT_ * H_;
    cvt_f32_bf16<<<dim3((n + 255) / 256), 256, 0, stream>>>(ht, ht_b, n);
    n = BATCH * TS_ * H_;
    cvt_f32_bf16<<<dim3((n + 255) / 256), 256, 0, stream>>>(hs, hs_b, n);
    // hs [TS][H] -> hs_t [H][TS], per batch
    transpose_cast<<<dim3(H_ / 32, TS_ / 32, BATCH), 256, 0, stream>>>(
        hs, hs_t, TS_, H_, (long long)TS_ * H_, (long long)H_ * TS_);
    // Wa [H][H] -> Wa_t [H][H]
    transpose_cast<<<dim3(H_ / 32, H_ / 32, 1), 256, 0, stream>>>(
        Wa, Wa_t, H_, H_, 0, 0);
    // Wc [2H][O] -> Wc_t [O][2H]
    transpose_cast<<<dim3(O_ / 32, (2 * H_) / 32, 1), 256, 0, stream>>>(
        Wc, Wc_t, 2 * H_, O_, 0, 0);
  }

  // 2) keys = hs @ W_a    (B_stored = Wa_t [N][K])       -> bf16 [B,TS,H]
  {
    GArgs g{hs_b, nullptr, (long long)TS_ * H_, H_,
            Wa_t, 0, H_,
            keys_b, (long long)TS_ * H_, H_,
            nullptr, H_, 0};
    wgemm_bf16<0, false><<<dim3(H_ / BN, TS_ / BM, BATCH), 256, 0, stream>>>(g);
  }

  // 3) scores = ht @ keys^T (B_stored = keys [TS][H])    -> f32 [B,TT,TS]
  {
    GArgs g{ht_b, nullptr, (long long)TT_ * H_, H_,
            keys_b, (long long)TS_ * H_, H_,
            sc_f, (long long)TT_ * TS_, TS_,
            nullptr, H_, 0};
    wgemm_bf16<1, false><<<dim3(TS_ / BN, TT_ / BM, BATCH), 256, 0, stream>>>(g);
  }

  // 4) masked softmax -> a (bf16)
  softmax_mask_kernel<<<dim3(TT_, BATCH), 256, 0, stream>>>(sc_f, src, a_b);

  // 5) c = a @ hs          (B_stored = hs_t [H][TS])     -> bf16 [B,TT,H]
  {
    GArgs g{a_b, nullptr, (long long)TT_ * TS_, TS_,
            hs_t, (long long)H_ * TS_, TS_,
            c_b, (long long)TT_ * H_, H_,
            nullptr, TS_, 0};
    wgemm_bf16<0, false><<<dim3(H_ / BN, TT_ / BM, BATCH), 256, 0, stream>>>(g);
  }

  // 6) out = tanh(concat(c, ht) @ W_c + b)  (B_stored = Wc_t [O][2H]) -> f32
  {
    GArgs g{c_b, ht_b, (long long)TT_ * H_, H_,
            Wc_t, 0, 2 * H_,
            out, (long long)TT_ * O_, O_,
            bias, 2 * H_, H_};
    wgemm_bf16<2, true><<<dim3(O_ / BN, TT_ / BM, BATCH), 256, 0, stream>>>(g);
  }
}
